// Pairs_Pruning_67095979099061
// MI455X (gfx1250) — matrix operand order repaired and checked by reading, not compiled, not run on hardware
//
#include <hip/hip_runtime.h>
#include <hip/hip_bf16.h>
#include <cmath>

typedef __attribute__((ext_vector_type(16))) _Float16 v16h;
typedef __attribute__((ext_vector_type(8)))  _Float16 v8h;
typedef __attribute__((ext_vector_type(8)))  float    v8f;

#define N_DET   20000
#define IN_DIM  151
#define KP1     160     // padded K for layer 1
#define HID     256     // 2*OUT_DIM
#define ODIM    128
#define N_EDGE  1000000

// ---------------- conversion kernels ----------------

// X (N_DET x IN_DIM f32) -> X16 (N_DET x KP1 f16, zero padded cols)
__global__ void cvt_x_kernel(const float* __restrict__ x, _Float16* __restrict__ xo, int total) {
  int i = blockIdx.x * blockDim.x + threadIdx.x;
  if (i >= total) return;
  int r = i / KP1, c = i % KP1;
  float v = (c < IN_DIM) ? x[r * IN_DIM + c] : 0.0f;
  xo[i] = (_Float16)v;
}

// dst[n*Kp + k] = (k<K) ? src[k*N + n] : 0  (transpose + pad + f32->f16)
__global__ void cvt_wT_kernel(const float* __restrict__ src, _Float16* __restrict__ dst,
                              int K, int N, int Kp) {
  int i = blockIdx.x * blockDim.x + threadIdx.x;
  if (i >= N * Kp) return;
  int n = i / Kp, k = i % Kp;
  float v = (k < K) ? src[k * N + n] : 0.0f;
  dst[i] = (_Float16)v;
}

// ---------------- WMMA fragment loaders ----------------
// A-matrix 16x32 f16 fragment (dense 16-bit A layout, ISA 7.12.2):
// lane half hi: K in [kc+hi*8, +8) and [kc+16+hi*8, +8)  -> two 16B runs
__device__ __forceinline__ v16h load_a_frag(const _Float16* rowPtr, int kc, int hi) {
  const v8h lo = *(const v8h*)(rowPtr + kc + hi * 8);
  const v8h hp = *(const v8h*)(rowPtr + kc + 16 + hi * 8);
  v16h a;
#pragma unroll
  for (int j = 0; j < 8; ++j) { a[j] = lo[j]; a[j + 8] = hp[j]; }
  return a;
}
// B-matrix 32x16 f16 fragment: lane half hi holds K in [kc+hi*8, +8) and
// [kc+16+hi*8, +8) of column n -> two 16B runs (matches A)
__device__ __forceinline__ v16h load_b_frag(const _Float16* colPtr, int kc, int hi) {
  const v8h lo = *(const v8h*)(colPtr + kc + hi * 8);
  const v8h hp = *(const v8h*)(colPtr + kc + 16 + hi * 8);
  v16h b;
#pragma unroll
  for (int j = 0; j < 8; ++j) { b[j] = lo[j]; b[j + 8] = hp[j]; }
  return b;
}

// ---------------- layer 1: H = relu(X @ W1 + b1), f16 out ----------------
// grid (1250, 2), block 128 (4 waves). wave w covers N in [w*64, w*64+64).
__global__ void __launch_bounds__(128)
gemm1_kernel(const _Float16* __restrict__ X,
             const _Float16* __restrict__ W1Ts, const float* __restrict__ b1s,
             const _Float16* __restrict__ W1To, const float* __restrict__ b1o,
             _Float16* __restrict__ Hs, _Float16* __restrict__ Ho) {
  const _Float16* WT = blockIdx.y ? W1To : W1Ts;
  const float*    bv = blockIdx.y ? b1o  : b1s;
  _Float16*       H  = blockIdx.y ? Ho   : Hs;

  const int wave = threadIdx.x >> 5;
  const int lane = threadIdx.x & 31;
  const int m    = lane & 15;
  const int hi   = lane >> 4;
  const int mBase = blockIdx.x * 16;
  const int nBase = wave * 64;

  const _Float16* pA = X + (size_t)(mBase + m) * KP1;
  v8f acc[4] = {};

#pragma unroll
  for (int kc = 0; kc < KP1; kc += 32) {
    v16h a = load_a_frag(pA, kc, hi);
#pragma unroll
    for (int t = 0; t < 4; ++t) {
      const _Float16* pB = WT + (size_t)(nBase + t * 16 + m) * KP1;
      v16h b = load_b_frag(pB, kc, hi);
      acc[t] = __builtin_amdgcn_wmma_f32_16x16x32_f16(
          false, a, false, b, (short)0, acc[t], false, false);
    }
  }

#pragma unroll
  for (int t = 0; t < 4; ++t) {
    const int n = nBase + t * 16 + m;
    const float bn = bv[n];
#pragma unroll
    for (int r = 0; r < 8; ++r) {
      float v = acc[t][r] + bn;
      v = v > 0.0f ? v : 0.0f;
      H[(size_t)(mBase + hi * 8 + r) * HID + n] = (_Float16)v;
    }
  }
}

// ---------------- layer 2: S = H @ W2 + b2, f32 out ----------------
// grid (1250, 2), block 64 (2 waves). wave w covers N in [w*64, w*64+64).
__global__ void __launch_bounds__(64)
gemm2_kernel(const _Float16* __restrict__ Hs, const _Float16* __restrict__ Ho,
             const _Float16* __restrict__ W2Ts, const float* __restrict__ b2s,
             const _Float16* __restrict__ W2To, const float* __restrict__ b2o,
             float* __restrict__ Ss, float* __restrict__ So) {
  const _Float16* A  = blockIdx.y ? Ho   : Hs;
  const _Float16* WT = blockIdx.y ? W2To : W2Ts;
  const float*    bv = blockIdx.y ? b2o  : b2s;
  float*          S  = blockIdx.y ? So   : Ss;

  const int wave = threadIdx.x >> 5;
  const int lane = threadIdx.x & 31;
  const int m    = lane & 15;
  const int hi   = lane >> 4;
  const int mBase = blockIdx.x * 16;
  const int nBase = wave * 64;

  const _Float16* pA = A + (size_t)(mBase + m) * HID;
  v8f acc[4] = {};

#pragma unroll
  for (int kc = 0; kc < HID; kc += 32) {
    v16h a = load_a_frag(pA, kc, hi);
#pragma unroll
    for (int t = 0; t < 4; ++t) {
      const _Float16* pB = WT + (size_t)(nBase + t * 16 + m) * HID;
      v16h b = load_b_frag(pB, kc, hi);
      acc[t] = __builtin_amdgcn_wmma_f32_16x16x32_f16(
          false, a, false, b, (short)0, acc[t], false, false);
    }
  }

#pragma unroll
  for (int t = 0; t < 4; ++t) {
    const int n = nBase + t * 16 + m;
    const float bn = bv[n];
#pragma unroll
    for (int r = 0; r < 8; ++r) {
      S[(size_t)(mBase + hi * 8 + r) * ODIM + n] = acc[t][r] + bn;
    }
  }
}

// ---------------- edge gather + dot + sigmoid ----------------
// one wave32 per edge: each lane loads float4 of sbj & obj rows (512B/row,
// fully coalesced, L2-resident), shuffle-reduce, lane 0 writes sigmoid.
__global__ void __launch_bounds__(256)
edge_dot_kernel(const long long* __restrict__ einds,
                const float* __restrict__ S, const float* __restrict__ O,
                float* __restrict__ out, int nE) {
  const int gtid = blockIdx.x * blockDim.x + threadIdx.x;
  const int wid  = gtid >> 5;
  const int lane = gtid & 31;
  if (wid >= nE) return;

  const long long i1 = einds[(size_t)wid * 3 + 1];
  const long long i2 = einds[(size_t)wid * 3 + 2];

  const float4 s4 = *(const float4*)(S + (size_t)i1 * ODIM + lane * 4);
  const float4 o4 = *(const float4*)(O + (size_t)i2 * ODIM + lane * 4);
  float p = s4.x * o4.x + s4.y * o4.y + s4.z * o4.z + s4.w * o4.w;

#pragma unroll
  for (int off = 16; off > 0; off >>= 1)
    p += __shfl_xor(p, off, 32);

  if (lane == 0)
    out[wid] = 1.0f / (1.0f + __expf(-p));
}

// ---------------- launch ----------------
extern "C" void kernel_launch(void* const* d_in, const int* in_sizes, int n_in,
                              void* d_out, int out_size, void* d_ws, size_t ws_size,
                              hipStream_t stream) {
  const float*     det_dists = (const float*)d_in[0];
  const long long* edge_inds = (const long long*)d_in[1];
  const float*     Ws_w1 = (const float*)d_in[2];
  const float*     Ws_b1 = (const float*)d_in[3];
  const float*     Ws_w2 = (const float*)d_in[4];
  const float*     Ws_b2 = (const float*)d_in[5];
  const float*     Wo_w1 = (const float*)d_in[6];
  const float*     Wo_b1 = (const float*)d_in[7];
  const float*     Wo_w2 = (const float*)d_in[8];
  const float*     Wo_b2 = (const float*)d_in[9];
  float* out = (float*)d_out;

  // workspace layout (all 16B aligned)
  char* ws = (char*)d_ws;
  _Float16* X16  = (_Float16*)ws;                       ws += (size_t)N_DET * KP1 * 2;   // 6.4 MB
  _Float16* W1sT = (_Float16*)ws;                       ws += (size_t)HID * KP1 * 2;     // 80 KB
  _Float16* W1oT = (_Float16*)ws;                       ws += (size_t)HID * KP1 * 2;
  _Float16* W2sT = (_Float16*)ws;                       ws += (size_t)ODIM * HID * 2;    // 64 KB
  _Float16* W2oT = (_Float16*)ws;                       ws += (size_t)ODIM * HID * 2;
  _Float16* Hs   = (_Float16*)ws;                       ws += (size_t)N_DET * HID * 2;   // 10.24 MB
  _Float16* Ho   = (_Float16*)ws;                       ws += (size_t)N_DET * HID * 2;
  float*    Ssb  = (float*)ws;                          ws += (size_t)N_DET * ODIM * 4;  // 10.24 MB
  float*    Sob  = (float*)ws;

  // 1) conversions
  {
    int total = N_DET * KP1;
    cvt_x_kernel<<<(total + 255) / 256, 256, 0, stream>>>(det_dists, X16, total);
  }
  cvt_wT_kernel<<<(HID * KP1 + 255) / 256, 256, 0, stream>>>(Ws_w1, W1sT, IN_DIM, HID, KP1);
  cvt_wT_kernel<<<(HID * KP1 + 255) / 256, 256, 0, stream>>>(Wo_w1, W1oT, IN_DIM, HID, KP1);
  cvt_wT_kernel<<<(ODIM * HID + 255) / 256, 256, 0, stream>>>(Ws_w2, W2sT, HID, ODIM, HID);
  cvt_wT_kernel<<<(ODIM * HID + 255) / 256, 256, 0, stream>>>(Wo_w2, W2oT, HID, ODIM, HID);

  // 2) layer 1 (both MLPs via grid.y)
  gemm1_kernel<<<dim3(N_DET / 16, 2), 128, 0, stream>>>(
      X16, W1sT, Ws_b1, W1oT, Wo_b1, Hs, Ho);

  // 3) layer 2
  gemm2_kernel<<<dim3(N_DET / 16, 2), 64, 0, stream>>>(
      Hs, Ho, W2sT, Ws_b2, W2oT, Wo_b2, Ssb, Sob);

  // 4) edge gather + dot + sigmoid (one wave per edge)
  {
    int waves = N_EDGE;
    int blocks = (waves * 32 + 255) / 256;
    edge_dot_kernel<<<blocks, 256, 0, stream>>>(edge_inds, Ssb, Sob, out, N_EDGE);
  }
}
